// DualScaleSTGAT_7533372637870
// MI455X (gfx1250) — compile-verified, matrix operand order checked by reading, not run on hardware
//
#include <hip/hip_runtime.h>
#include <cmath>

// ---------------- WMMA types / helpers (gfx1250, wave32) ----------------
typedef __attribute__((ext_vector_type(16))) _Float16 v16h;
typedef __attribute__((ext_vector_type(8)))  float    v8f;

__device__ __forceinline__ float gelu_f(float v) {
  return 0.5f * v * (1.0f + erff(v * 0.70710678118654752f));
}
__device__ __forceinline__ float sigmoid_f(float v) {
  return 1.0f / (1.0f + expf(-v));
}
// order-preserving float -> unsigned map (for atomic segment-max)
__device__ __forceinline__ unsigned enc_f(float f) {
  int i = __float_as_int(f);
  unsigned u = (unsigned)i;
  return (i < 0) ? ~u : (u | 0x80000000u);
}
__device__ __forceinline__ float dec_f(unsigned e) {
  unsigned u = (e & 0x80000000u) ? (e & 0x7FFFFFFFu) : ~e;
  return __uint_as_float(u);
}

// ---- fragment loader (branch-free; K-guard folds at compile time) ----
// ISA 16-bit A layout (16x32): lane(0-15)=row; slot j covers
//   k = k0 + (j>>2)*16 + (lane>=16 ? 8 : 0) + (j&3)*2   (pairs)
// For K a multiple of 16, slot validity is independent of the lane half:
//   slot j valid  <=>  k0 + ((j>>2)+1)*16 <= K
// Row must be pre-clamped to valid memory; strides are even (8B-aligned float2).
template <int K>
__device__ __forceinline__ v16h frag_ld(const float* P, int ld, int row, int k0) {
  int hi8 = ((threadIdx.x >> 4) & 1) << 3;
  const float* p = P + (size_t)row * ld + k0 + hi8;
  v16h r;
#pragma unroll
  for (int j = 0; j < 8; ++j) {
    int koff = ((j >> 2) << 4) + ((j & 3) << 1);
    if (k0 + ((j >> 2) + 1) * 16 <= K) {   // constant-folds after unroll
      float2 f = *(const float2*)(p + koff);
      r[2 * j]     = (_Float16)f.x;
      r[2 * j + 1] = (_Float16)f.y;
    } else {
      r[2 * j]     = (_Float16)0.f;
      r[2 * j + 1] = (_Float16)0.f;
    }
  }
  return r;
}

// ---------------- fully-specialized GEMM: C[M,NOUT] = act(A[M,K] @ W^T + bias (+C)) ----
// one wave per 16x(16*TILES) strip; A fragment reused across TILES WMMAs.
// All B fragments are materialized before the WMMA burst so the TILES WMMAs
// have disjoint A/B/D registers (no WMMA->VALU WAR nops between them).
template <int K, int NOUT, int ACT, int ACCUM>
__global__ __launch_bounds__(32)
void gemm_tpl(const float* __restrict__ A, int lda,
              const float* __restrict__ W, int ldw,
              const float* __restrict__ bias,
              float* __restrict__ C, int M) {
  constexpr int TILES = (NOUT % 64 == 0) ? 4 : ((NOUT % 32 == 0) ? 2 : 1);
  constexpr int KT = (K + 31) / 32;
  constexpr bool RAGN = (NOUT % 16) != 0;  // only NOUT=4 here
  int m0 = blockIdx.x * 16;
  int nbase = blockIdx.y * (16 * TILES);
  int lane = threadIdx.x & 31;
  int l15 = lane & 15;
  int rowA = m0 + l15;
  rowA = (rowA < M) ? rowA : (M - 1);  // clamp: garbage rows discarded at store

  v8f acc[TILES];
#pragma unroll
  for (int t = 0; t < TILES; ++t) acc[t] = v8f{};

#pragma unroll
  for (int kt = 0; kt < KT; ++kt) {
    int k0 = kt * 32;
    v16h a = frag_ld<K>(A, lda, rowA, k0);
    v16h b[TILES];
#pragma unroll
    for (int t = 0; t < TILES; ++t) {
      int rn = nbase + t * 16 + l15;
      if (RAGN) rn = (rn < NOUT) ? rn : (NOUT - 1);
      b[t] = frag_ld<K>(W, ldw, rn, k0);
    }
#pragma unroll
    for (int t = 0; t < TILES; ++t)
      acc[t] = __builtin_amdgcn_wmma_f32_16x16x32_f16(false, a, false, b[t], (short)0,
                                                      acc[t], false, false);
  }

  int mh = (lane >> 4) << 3;  // lanes 16-31 hold rows +8
  bool mfull = (m0 + 16 <= M);
#pragma unroll
  for (int t = 0; t < TILES; ++t) {
    int n = nbase + t * 16 + l15;
    if (!RAGN || n < NOUT) {
      float bv = bias ? bias[n] : 0.f;
      if (mfull) {
#pragma unroll
        for (int r = 0; r < 8; ++r) {
          size_t o = (size_t)(m0 + mh + r) * NOUT + n;
          float v = acc[t][r] + bv;
          if (ACCUM) v += C[o];
          if (ACT == 1) v = gelu_f(v);
          if (ACT == 2) v = sigmoid_f(v);
          C[o] = v;
        }
      } else {
#pragma unroll
        for (int r = 0; r < 8; ++r) {
          int m = m0 + mh + r;
          if (m < M) {
            size_t o = (size_t)m * NOUT + n;
            float v = acc[t][r] + bv;
            if (ACCUM) v += C[o];
            if (ACT == 1) v = gelu_f(v);
            if (ACT == 2) v = sigmoid_f(v);
            C[o] = v;
          }
        }
      }
    }
  }
}

// ---------------- LayerNorm (+optional pre-add) -> GELU (+optional post-add) ----------------
// one wave per row; D in {64,96,128}
__global__ __launch_bounds__(32)
void ln_act_kernel(const float* __restrict__ in, const float* __restrict__ addpre,
                   const float* __restrict__ addpost,
                   const float* __restrict__ g, const float* __restrict__ b,
                   float* __restrict__ out, int M, int D) {
  int row = blockIdx.x;
  if (row >= M) return;
  int lane = threadIdx.x;
  int per = D >> 5;
  size_t base = (size_t)row * D;
  float vals[4];
  float s = 0.f;
#pragma unroll
  for (int i = 0; i < 4; ++i)
    if (i < per) {
      int idx = i * 32 + lane;
      float v = in[base + idx];
      if (addpre) v += addpre[base + idx];
      vals[i] = v;
      s += v;
    }
  for (int off = 16; off > 0; off >>= 1) s += __shfl_xor(s, off, 32);
  float mean = s / (float)D;
  float vs = 0.f;
#pragma unroll
  for (int i = 0; i < 4; ++i)
    if (i < per) { float d = vals[i] - mean; vs += d * d; }
  for (int off = 16; off > 0; off >>= 1) vs += __shfl_xor(vs, off, 32);
  float inv = rsqrtf(vs / (float)D + 1e-5f);
#pragma unroll
  for (int i = 0; i < 4; ++i)
    if (i < per) {
      int idx = i * 32 + lane;
      float y = (vals[i] - mean) * inv * g[idx] + b[idx];
      y = gelu_f(y);
      if (addpost) y += addpost[base + idx];
      out[base + idx] = y;
    }
}

// ---------------- GRU: whole time loop in-kernel, 16-node tile per 1-wave block ----------------
// h in LDS; gh = h@Whh^T via WMMA; (layer with DIN>1: gx = x_t@Wih^T via WMMA)
template <int H, int DIN, bool WRITE_SEQ>
__global__ __launch_bounds__(32)
void gru_kernel(const float* __restrict__ xin, int x_ld, int x_off,   // DIN==1 path
                const float* __restrict__ seqin,                      // DIN>1: (N,T,DIN)
                const float* __restrict__ Wih, const float* __restrict__ Whh,
                const float* __restrict__ bih, const float* __restrict__ bhh,
                float* __restrict__ seqout, float* __restrict__ hout,
                int Nn, int T) {
  constexpr int G = 3 * H;
  __shared__ float hbuf[16 * H];
  __shared__ float gxb[16 * G];
  __shared__ float ghb[16 * G];
  int m0 = blockIdx.x * 16;
  int lane = threadIdx.x;
  int l15 = lane & 15;
  int mh = (lane >> 4) << 3;
  int rowA = m0 + l15;
  rowA = (rowA < Nn) ? rowA : (Nn - 1);  // clamp (pad rows never stored)
  for (int i = lane; i < 16 * H; i += 32) hbuf[i] = 0.f;
  __syncthreads();

  for (int t = 0; t < T; ++t) {
    // ---- gx = x_t @ Wih^T + bih ----
    if constexpr (DIN == 1) {
      for (int i = lane; i < 16 * G; i += 32) {
        int m = i / G, j = i % G;
        int node = m0 + m;
        node = (node < Nn) ? node : (Nn - 1);  // clamp
        float xv = xin[(size_t)node * x_ld + x_off + t];
        gxb[i] = xv * Wih[j] + bih[j];
      }
    } else {
      const float* At = seqin + (size_t)t * DIN;  // row stride T*DIN (even)
      for (int nt = 0; nt < G / 16; ++nt) {
        v8f c = {};
#pragma unroll
        for (int k0 = 0; k0 < DIN; k0 += 32) {
          v16h a = frag_ld<DIN>(At, T * DIN, rowA, k0);
          v16h b = frag_ld<DIN>(Wih, DIN, nt * 16 + l15, k0);
          c = __builtin_amdgcn_wmma_f32_16x16x32_f16(false, a, false, b, (short)0,
                                                     c, false, false);
        }
        int n = nt * 16 + l15;
#pragma unroll
        for (int r = 0; r < 8; ++r) gxb[(mh + r) * G + n] = c[r] + bih[n];
      }
    }
    __syncthreads();
    // ---- gh = h @ Whh^T + bhh ----  (LDS h tile is always full 16 x H)
    for (int nt = 0; nt < G / 16; ++nt) {
      v8f c = {};
#pragma unroll
      for (int k0 = 0; k0 < H; k0 += 32) {
        v16h a = frag_ld<H>(hbuf, H, l15, k0);
        v16h b = frag_ld<H>(Whh, H, nt * 16 + l15, k0);
        c = __builtin_amdgcn_wmma_f32_16x16x32_f16(false, a, false, b, (short)0, c,
                                                   false, false);
      }
      int n = nt * 16 + l15;
#pragma unroll
      for (int r = 0; r < 8; ++r) ghb[(mh + r) * G + n] = c[r] + bhh[n];
    }
    __syncthreads();
    // ---- gates ----
    for (int i = lane; i < 16 * H; i += 32) {
      int m = i / H, j = i % H;
      float r = sigmoid_f(gxb[m * G + j] + ghb[m * G + j]);
      float z = sigmoid_f(gxb[m * G + H + j] + ghb[m * G + H + j]);
      float nn = tanhf(gxb[m * G + 2 * H + j] + r * ghb[m * G + 2 * H + j]);
      float hv = (1.f - z) * nn + z * hbuf[i];
      hbuf[i] = hv;
      if constexpr (WRITE_SEQ) {
        int node = m0 + m;
        if (node < Nn) seqout[(size_t)node * T * H + (size_t)t * H + j] = hv;
      }
    }
    __syncthreads();
  }
  for (int i = lane; i < 16 * H; i += 32) {
    int m = i / H, j = i % H;
    int node = m0 + m;
    if (node < Nn) hout[(size_t)node * H + j] = hbuf[i];
  }
}

// ---------------- element-wise / graph kernels ----------------
__global__ void fill_u32_kernel(unsigned* p, unsigned v, long n) {
  long i = (long)blockIdx.x * blockDim.x + threadIdx.x;
  if (i < n) p[i] = v;
}

__global__ void edge_count_sum_kernel(const long long* __restrict__ dst,
                                      const float* __restrict__ eattr,
                                      float* __restrict__ cnt, float* __restrict__ easum,
                                      int E) {
  int e = blockIdx.x * blockDim.x + threadIdx.x;
  if (e >= E) return;
  int d = (int)dst[e];
  atomicAdd(&cnt[d], 1.f);
#pragma unroll
  for (int k = 0; k < 4; ++k) atomicAdd(&easum[d * 4 + k], eattr[(size_t)e * 4 + k]);
}

__global__ void ea_loop_div_kernel(const float* __restrict__ cnt,
                                   const float* __restrict__ easum,
                                   float* __restrict__ ealoop, int Nn) {
  int idx = blockIdx.x * blockDim.x + threadIdx.x;
  if (idx >= Nn * 4) return;
  ealoop[idx] = easum[idx] / fmaxf(cnt[idx / 4], 1.f);
}

// per (edge, head): logit = sum_c att[h,c]*leakyrelu(xl[s,h,c]+xr[d,h,c]+eattr@We)
__global__ void gat_logits_kernel(const float* __restrict__ xl, const float* __restrict__ xr,
                                  const long long* __restrict__ src,
                                  const long long* __restrict__ dst,
                                  const float* __restrict__ eattr,
                                  const float* __restrict__ ealoop,
                                  const float* __restrict__ We, const float* __restrict__ att,
                                  float* __restrict__ logits, unsigned* __restrict__ mx,
                                  int E, int Nn, int heads, int C) {
  int idx = blockIdx.x * blockDim.x + threadIdx.x;
  int total = (E + Nn) * heads;
  if (idx >= total) return;
  int e = idx / heads, h = idx % heads;
  int s, d;
  const float* ea;
  if (e < E) { s = (int)src[e]; d = (int)dst[e]; ea = eattr + (size_t)e * 4; }
  else       { s = d = e - E;                    ea = ealoop + (size_t)(e - E) * 4; }
  float e0 = ea[0], e1 = ea[1], e2 = ea[2], e3 = ea[3];
  const float* xls = xl + (size_t)s * heads * C + (size_t)h * C;
  const float* xrd = xr + (size_t)d * heads * C + (size_t)h * C;
  const float* ah  = att + (size_t)h * C;
  const float* wh  = We + (size_t)h * C * 4;
  float acc = 0.f;
  for (int c = 0; c < C; ++c) {
    const float* w = wh + (size_t)c * 4;
    float ep = e0 * w[0] + e1 * w[1] + e2 * w[2] + e3 * w[3];
    float m = xls[c] + xrd[c] + ep;
    m = (m > 0.f) ? m : 0.2f * m;  // leaky_relu 0.2
    acc += ah[c] * m;
  }
  logits[idx] = acc;
  atomicMax(&mx[d * heads + h], enc_f(acc));
}

__global__ void gat_alpha_kernel(float* __restrict__ logits, const unsigned* __restrict__ mx,
                                 float* __restrict__ ssum, const long long* __restrict__ dst,
                                 int E, int Nn, int heads) {
  int idx = blockIdx.x * blockDim.x + threadIdx.x;
  int total = (E + Nn) * heads;
  if (idx >= total) return;
  int e = idx / heads, h = idx % heads;
  int d = (e < E) ? (int)dst[e] : (e - E);
  float a = expf(logits[idx] - dec_f(mx[d * heads + h]));
  logits[idx] = a;
  atomicAdd(&ssum[d * heads + h], a);
}

__global__ void gat_scatter_kernel(const float* __restrict__ xl, const float* __restrict__ alpha,
                                   const float* __restrict__ ssum,
                                   const long long* __restrict__ src,
                                   const long long* __restrict__ dst,
                                   float* __restrict__ out, int E, int Nn, int heads, int C) {
  int idx = blockIdx.x * blockDim.x + threadIdx.x;
  int total = (E + Nn) * heads;
  if (idx >= total) return;
  int e = idx / heads, h = idx % heads;
  int s, d;
  if (e < E) { s = (int)src[e]; d = (int)dst[e]; }
  else       { s = d = e - E; }
  float a = alpha[idx] / ssum[d * heads + h];
  const float* xls = xl + (size_t)s * heads * C + (size_t)h * C;
  float* od = out + (size_t)d * heads * C + (size_t)h * C;
  for (int c = 0; c < C; ++c) atomicAdd(&od[c], a * xls[c]);
}

// pre = mean_h(gatout) + gat_bias + resbuf
__global__ void gat_finalize_kernel(const float* __restrict__ gatout,
                                    const float* __restrict__ gbias,
                                    const float* __restrict__ resbuf,
                                    float* __restrict__ pre, int Nn, int heads, int C) {
  int idx = blockIdx.x * blockDim.x + threadIdx.x;
  if (idx >= Nn * C) return;
  int i = idx / C, c = idx % C;
  float s = 0.f;
  for (int h = 0; h < heads; ++h) s += gatout[(size_t)i * heads * C + (size_t)h * C + c];
  pre[idx] = s / (float)heads + gbias[c] + resbuf[idx];
}

__global__ void sage_max_kernel(const long long* __restrict__ src,
                                const long long* __restrict__ dst,
                                const float* __restrict__ h2, unsigned* __restrict__ encv,
                                int E, int C) {
  int idx = blockIdx.x * blockDim.x + threadIdx.x;
  if (idx >= E * C) return;
  int e = idx / C, c = idx % C;
  atomicMax(&encv[(size_t)dst[e] * C + c], enc_f(h2[(size_t)src[e] * C + c]));
}

__global__ void sage_decode_kernel(const unsigned* __restrict__ encv, float* __restrict__ agg,
                                   long n) {
  long i = (long)blockIdx.x * blockDim.x + threadIdx.x;
  if (i >= n) return;
  unsigned e = encv[i];
  agg[i] = (e != 0u) ? dec_f(e) : 0.f;  // untouched segments -> 0 (isfinite filter)
}

// ---------------- host-side launch helpers ----------------
template <int K, int NOUT, int ACT, int ACCUM>
static void launch_gemm_t(const float* A, int lda, const float* W, int ldw,
                          const float* bias, float* C, int M, hipStream_t s) {
  constexpr int TILES = (NOUT % 64 == 0) ? 4 : ((NOUT % 32 == 0) ? 2 : 1);
  dim3 grid((M + 15) / 16, (NOUT + 16 * TILES - 1) / (16 * TILES));
  gemm_tpl<K, NOUT, ACT, ACCUM><<<grid, 32, 0, s>>>(A, lda, W, ldw, bias, C, M);
}
static void launch_fill(void* p, unsigned v, long n, hipStream_t s) {
  fill_u32_kernel<<<(unsigned)((n + 255) / 256), 256, 0, s>>>((unsigned*)p, v, n);
}
static inline int ceil_div(long a, long b) { return (int)((a + b - 1) / b); }

extern "C" void kernel_launch(void* const* d_in, const int* in_sizes, int n_in,
                              void* d_out, int out_size, void* d_ws, size_t ws_size,
                              hipStream_t stream) {
  const float* x = (const float*)d_in[0];
  const long long* ei = (const long long*)d_in[1];
  const float* eattr = (const float*)d_in[2];
  auto P = [&](int i) { return (const float*)d_in[i]; };

  const int N = in_sizes[0] / 34;
  const int E = in_sizes[1] / 2;
  const int F = E + N;  // edges + self loops
  const long long* src = ei;
  const long long* dst = ei + E;

  // ---- workspace carve-up ----
  float* ws = (float*)d_ws;
  size_t cur = 0;
  auto alloc = [&](size_t nfl) { float* p = ws + cur; cur += nfl; return p; };
  float* t128    = alloc((size_t)N * 128);
  float* skip64  = alloc((size_t)N * 64);
  float* t64     = alloc((size_t)N * 64);
  float* s64     = alloc((size_t)N * 64);
  float* seq1    = alloc((size_t)N * 6 * 96);
  float* hshort  = alloc((size_t)N * 96);
  float* hlong   = alloc((size_t)N * 64);
  float* ht96pre = alloc((size_t)N * 96);
  float* htemp   = alloc((size_t)N * 96);
  float* fupre   = alloc((size_t)N * 128);
  float* fused   = alloc((size_t)N * 128);
  float* cnt     = alloc((size_t)N);
  float* easum   = alloc((size_t)N * 4);
  float* ealoop  = alloc((size_t)N * 4);
  float* xl1     = alloc((size_t)N * 1024);
  float* xr1     = alloc((size_t)N * 1024);
  float* logits1 = alloc((size_t)F * 8);
  float* mx1f    = alloc((size_t)N * 8);
  float* sum1    = alloc((size_t)N * 8);
  float* res1buf = alloc((size_t)N * 128);
  float* h1pre   = alloc((size_t)N * 128);
  float* h1      = alloc((size_t)N * 128);
  float* xl2     = alloc((size_t)N * 256);
  float* xr2     = alloc((size_t)N * 256);
  float* logits2 = alloc((size_t)F * 4);
  float* mx2f    = alloc((size_t)N * 4);
  float* sum2    = alloc((size_t)N * 4);
  float* res2buf = alloc((size_t)N * 64);
  float* h2pre   = alloc((size_t)N * 64);
  float* h2b     = alloc((size_t)N * 64);
  float* sageenc = alloc((size_t)N * 64);
  float* agg     = alloc((size_t)N * 64);
  float* hspre   = alloc((size_t)N * 64);
  float* h32     = alloc((size_t)N * 32);
  // big accumulators alias xr buffers (safe: used strictly after logits pass)
  float* gatout1 = xr1;
  float* gatout2 = xr2;

  const int TB = 256;

  // ---- self-loop edge-attr means (used by both GAT layers) ----
  launch_fill(cnt, 0u, N, stream);
  launch_fill(easum, 0u, (long)N * 4, stream);
  edge_count_sum_kernel<<<ceil_div(E, TB), TB, 0, stream>>>(dst, eattr, cnt, easum, E);
  ea_loop_div_kernel<<<ceil_div((long)N * 4, TB), TB, 0, stream>>>(cnt, easum, ealoop, N);

  // ---- static branch: 16 -> 128 -> 64 (+skip) ----
  launch_gemm_t<16, 128, 0, 0>(x, 34, P(3), 16, P(4), t128, N, stream);
  ln_act_kernel<<<N, 32, 0, stream>>>(t128, nullptr, nullptr, P(5), P(6), t128, N, 128);
  launch_gemm_t<128, 64, 0, 0>(t128, 128, P(7), 128, P(8), t64, N, stream);
  launch_gemm_t<16, 64, 0, 0>(x, 34, P(11), 16, nullptr, skip64, N, stream);
  ln_act_kernel<<<N, 32, 0, stream>>>(t64, nullptr, skip64, P(9), P(10), s64, N, 64);

  // ---- GRUs (WMMA recurrence in-kernel) ----
  gru_kernel<96, 1, true><<<ceil_div(N, 16), 32, 0, stream>>>(
      x, 34, 16, nullptr, P(12), P(13), P(14), P(15), seq1, hshort, N, 6);
  gru_kernel<96, 96, false><<<ceil_div(N, 16), 32, 0, stream>>>(
      nullptr, 0, 0, seq1, P(16), P(17), P(18), P(19), nullptr, hshort, N, 6);
  gru_kernel<64, 1, false><<<ceil_div(N, 16), 32, 0, stream>>>(
      x, 34, 22, nullptr, P(20), P(21), P(22), P(23), nullptr, hlong, N, 12);

  // ---- gate: [hshort|hlong] (160) -> 96, LN, GELU ----
  launch_gemm_t<96, 96, 0, 0>(hshort, 96, P(24), 160, P(25), ht96pre, N, stream);
  launch_gemm_t<64, 96, 0, 1>(hlong, 64, P(24) + 96, 160, nullptr, ht96pre, N, stream);
  ln_act_kernel<<<N, 32, 0, stream>>>(ht96pre, nullptr, nullptr, P(26), P(27), htemp, N, 96);

  // ---- fusion: [s|htemp] (160) -> 128, LN, GELU ----
  launch_gemm_t<64, 128, 0, 0>(s64, 64, P(28), 160, P(29), fupre, N, stream);
  launch_gemm_t<96, 128, 0, 1>(htemp, 96, P(28) + 64, 160, nullptr, fupre, N, stream);
  ln_act_kernel<<<N, 32, 0, stream>>>(fupre, nullptr, nullptr, P(30), P(31), fused, N, 128);

  // ---- GATv2 layer 1: heads=8, C=128 ----
  launch_gemm_t<128, 1024, 0, 0>(fused, 128, P(32), 128, P(33), xl1, N, stream);
  launch_gemm_t<128, 1024, 0, 0>(fused, 128, P(34), 128, P(35), xr1, N, stream);
  launch_fill(mx1f, 0u, (long)N * 8, stream);
  launch_fill(sum1, 0u, (long)N * 8, stream);
  gat_logits_kernel<<<ceil_div((long)F * 8, TB), TB, 0, stream>>>(
      xl1, xr1, src, dst, eattr, ealoop, P(36), P(37), logits1, (unsigned*)mx1f, E, N, 8, 128);
  launch_fill(gatout1, 0u, (long)N * 1024, stream);  // reuses xr1 (done with it)
  gat_alpha_kernel<<<ceil_div((long)F * 8, TB), TB, 0, stream>>>(
      logits1, (unsigned*)mx1f, sum1, dst, E, N, 8);
  gat_scatter_kernel<<<ceil_div((long)F * 8, TB), TB, 0, stream>>>(
      xl1, logits1, sum1, src, dst, gatout1, E, N, 8, 128);
  launch_gemm_t<128, 128, 0, 0>(fused, 128, P(41), 128, nullptr, res1buf, N, stream);
  gat_finalize_kernel<<<ceil_div((long)N * 128, TB), TB, 0, stream>>>(
      gatout1, P(38), res1buf, h1pre, N, 8, 128);
  ln_act_kernel<<<N, 32, 0, stream>>>(h1pre, nullptr, nullptr, P(39), P(40), h1, N, 128);

  // ---- GATv2 layer 2: heads=4, C=64 ----
  launch_gemm_t<128, 256, 0, 0>(h1, 128, P(42), 128, P(43), xl2, N, stream);
  launch_gemm_t<128, 256, 0, 0>(h1, 128, P(44), 128, P(45), xr2, N, stream);
  launch_fill(mx2f, 0u, (long)N * 4, stream);
  launch_fill(sum2, 0u, (long)N * 4, stream);
  gat_logits_kernel<<<ceil_div((long)F * 4, TB), TB, 0, stream>>>(
      xl2, xr2, src, dst, eattr, ealoop, P(46), P(47), logits2, (unsigned*)mx2f, E, N, 4, 64);
  launch_fill(gatout2, 0u, (long)N * 256, stream);  // reuses xr2
  gat_alpha_kernel<<<ceil_div((long)F * 4, TB), TB, 0, stream>>>(
      logits2, (unsigned*)mx2f, sum2, dst, E, N, 4);
  gat_scatter_kernel<<<ceil_div((long)F * 4, TB), TB, 0, stream>>>(
      xl2, logits2, sum2, src, dst, gatout2, E, N, 4, 64);
  launch_gemm_t<128, 64, 0, 0>(h1, 128, P(51), 128, nullptr, res2buf, N, stream);
  gat_finalize_kernel<<<ceil_div((long)N * 64, TB), TB, 0, stream>>>(
      gatout2, P(48), res2buf, h2pre, N, 4, 64);
  ln_act_kernel<<<N, 32, 0, stream>>>(h2pre, nullptr, nullptr, P(49), P(50), h2b, N, 64);

  // ---- SAGE max ----
  launch_fill(sageenc, 0u, (long)N * 64, stream);
  sage_max_kernel<<<ceil_div((long)E * 64, TB), TB, 0, stream>>>(
      src, dst, h2b, (unsigned*)sageenc, E, 64);
  sage_decode_kernel<<<ceil_div((long)N * 64, TB), TB, 0, stream>>>(
      (unsigned*)sageenc, agg, (long)N * 64);
  launch_gemm_t<64, 64, 0, 0>(agg, 64, P(52), 64, P(53), hspre, N, stream);
  launch_gemm_t<64, 64, 0, 1>(h2b, 64, P(54), 64, nullptr, hspre, N, stream);
  ln_act_kernel<<<N, 32, 0, stream>>>(hspre, h2b, nullptr, P(55), P(56), hspre, N, 64);

  // ---- head: 64 -> 32 (GELU) -> 4 (sigmoid) ----
  launch_gemm_t<64, 32, 1, 0>(hspre, 64, P(57), 64, P(58), h32, N, stream);
  launch_gemm_t<32, 4, 2, 0>(h32, 32, P(59), 32, P(60), (float*)d_out, N, stream);

  (void)n_in; (void)out_size; (void)ws_size;
}
